// SpatioTemporalImputer_63969242906779
// MI455X (gfx1250) — matrix operand order repaired and checked by reading, not compiled, not run on hardware
//
#include <hip/hip_runtime.h>
#include <hip/hip_bf16.h>
#include <math.h>

#define cB 32
#define cN 256
#define cI 16
#define cO 16
#define cCG 64
#define cH 64
#define cE 4096

typedef __attribute__((ext_vector_type(16))) _Float16 v16h;
typedef __attribute__((ext_vector_type(4)))  _Float16 v4h;
typedef __attribute__((ext_vector_type(8)))  float    v8f;

// ---------------------------------------------------------------------------
// Batched GEMM via v_wmma_f32_16x16x32_f16, double-buffered LDS.
// block = 256 threads (8 waves), block tile = 32(M) x 64(N); each wave owns one
// 16x16 C tile. Per K-step: issue next tile's wide global loads, WMMA on the
// current LDS buffer (hides HBM latency), convert+store into the other buffer,
// single barrier. Fragments are stored in WMMA fragment order as f16 so reads
// are contiguous 128-bit LDS loads.
// Requirements: M % 32 == 0, N % 64 == 0, K % 32 == 0.
// TRANSB==0: B is [K,N] row-major.  TRANSB==1: B is [N,K] row-major (A*B^T).
// grid = (N/64, M/32, batch).
// ---------------------------------------------------------------------------
template <int TRANSB>
__global__ __launch_bounds__(256)
void gemm_wmma_kernel(const float* __restrict__ A, const float* __restrict__ Bm,
                      float* __restrict__ C, int M, int N, int K,
                      long long strideA, long long strideB, long long strideC,
                      float alpha, const float* __restrict__ bias, int act)
{
    __shared__ alignas(32) _Float16 AsH[2][2][32][16];   // [buf][mt][lane][elem]
    __shared__ alignas(32) _Float16 BsH[2][4][32][16];   // [buf][nt][lane][elem]
    const int ABUF = 2 * 32 * 16;
    const int BBUF = 4 * 32 * 16;

    const int tid  = threadIdx.x;
    const int lane = tid & 31;
    const int wv   = tid >> 5;          // 0..7
    const int wmt  = wv >> 2;           // wave's M sub-tile (0..1)
    const int wnt  = wv & 3;            // wave's N sub-tile (0..3)
    const int tn = blockIdx.x, tm = blockIdx.y, bz = blockIdx.z;

    const float* Ab = A + (long long)bz * strideA + (long long)tm * 32 * K;
    const float* Bb = Bm + (long long)bz * strideB;
    float* Cb = C + (long long)bz * strideC;

    // ---- A staging geometry: one float4 per thread at (ar, ac..ac+3) ----
    const int ar = tid >> 3;                 // 0..31
    const int ac = (tid & 7) * 4;            // 0,4,...,28
    const int ahh = ac >> 4;
    const int ael = ac & 15;     // 4 contiguous elems
    _Float16* abase = &AsH[0][ar >> 4][(ar & 15) + 16 * ahh][ael];
    const float* asrc = Ab + (long long)ar * K + ac;

    // ---- B staging geometry: two 4-element groups per thread ----
    int bn0, bk0, bn1, bk1;
    if (TRANSB) {
        int l0 = tid * 4;        bn0 = l0 >> 5; bk0 = l0 & 31;   // 4 consecutive k
        int l1 = tid * 4 + 1024; bn1 = l1 >> 5; bk1 = l1 & 31;
    } else {
        bn0 = bn1 = tid & 63;                                    // fixed column
        bk0 = (tid >> 6) * 4;                                    // 0,4,8,12
        bk1 = bk0 + 16;
    }
    _Float16* bbase0 = &BsH[0][bn0 >> 4][(bn0 & 15) + 16 * (bk0 >> 4)][bk0 & 15];
    _Float16* bbase1 = &BsH[0][bn1 >> 4][(bn1 & 15) + 16 * (bk1 >> 4)][bk1 & 15];

    const int m16 = lane & 15, half = lane >> 4;
    v8f acc = {0.f, 0.f, 0.f, 0.f, 0.f, 0.f, 0.f, 0.f};

    float4 av, b0, b1;

    // ---- tile fetch into registers ----
    auto fetch = [&](int kk) {
        av = *(const float4*)(asrc + kk);
        if (TRANSB) {
            b0 = *(const float4*)(Bb + (long long)(tn * 64 + bn0) * K + kk + bk0);
            b1 = *(const float4*)(Bb + (long long)(tn * 64 + bn1) * K + kk + bk1);
        } else {
            const float* p0 = Bb + (long long)(kk + bk0) * N + tn * 64 + bn0;
            const float* p1 = Bb + (long long)(kk + bk1) * N + tn * 64 + bn0;
            b0.x = p0[0]; b0.y = p0[N]; b0.z = p0[2 * (long long)N]; b0.w = p0[3 * (long long)N];
            b1.x = p1[0]; b1.y = p1[N]; b1.z = p1[2 * (long long)N]; b1.w = p1[3 * (long long)N];
        }
    };
    // ---- convert + 64-bit LDS stores into buffer `buf` ----
    auto stage = [&](int buf) {
        v4h ha4 = {(_Float16)av.x, (_Float16)av.y, (_Float16)av.z, (_Float16)av.w};
        v4h hb0 = {(_Float16)b0.x, (_Float16)b0.y, (_Float16)b0.z, (_Float16)b0.w};
        v4h hb1 = {(_Float16)b1.x, (_Float16)b1.y, (_Float16)b1.z, (_Float16)b1.w};
        *(v4h*)(abase + buf * ABUF)   = ha4;
        *(v4h*)(bbase0 + buf * BBUF)  = hb0;
        *(v4h*)(bbase1 + buf * BBUF)  = hb1;
    };
    // ---- contiguous 128-bit fragment loads + WMMA on buffer `buf` ----
    auto mma = [&](int buf) {
        const v16h a = *(const v16h*)(&AsH[buf][wmt][lane][0]);
        const v16h b = *(const v16h*)(&BsH[buf][wnt][lane][0]);
        acc = __builtin_amdgcn_wmma_f32_16x16x32_f16(false, a, false, b,
                                                     (short)0, acc, false, false);
    };

    // prologue: fill buffer 0
    fetch(0);
    stage(0);
    __syncthreads();

    int cur = 0;
    for (int kk = 32; kk < K; kk += 32) {
        fetch(kk);                        // next tile's loads in flight
        if (kk + 32 < K) {
            const void* pf = TRANSB
                ? (const void*)(Bb + (long long)(tn * 64 + bn0) * K + kk + 32 + bk0)
                : (const void*)(Bb + (long long)(kk + 32 + bk0) * N + tn * 64 + bn0);
            __builtin_prefetch(pf, 0, 1);
        }
        mma(cur);                         // compute on current buffer
        stage(cur ^ 1);                   // store next tile into other buffer
        __syncthreads();
        cur ^= 1;
    }
    mma(cur);                             // epilogue

    #pragma unroll
    for (int r = 0; r < 8; ++r) {
        int row = tm * 32 + wmt * 16 + r + 8 * half;
        int col = tn * 64 + wnt * 16 + m16;
        float v = acc[r] * alpha;
        if (bias) v += bias[col];
        if (act == 1) v = fmaxf(v, 0.f);
        Cb[(long long)row * N + col] = v;
    }
}

// ---------------------------------------------------------------------------
// Small helper kernels
// ---------------------------------------------------------------------------
__global__ void zero_f32(float* p, long long n) {
    long long i = (long long)blockIdx.x * blockDim.x + threadIdx.x;
    long long st = (long long)gridDim.x * blockDim.x;
    for (; i < n; i += st) p[i] = 0.f;
}

__global__ void copy_f32(const float* __restrict__ s, float* __restrict__ d, int n) {
    int i = blockIdx.x * blockDim.x + threadIdx.x;
    if (i < n) d[i] = s[i];
}

// masked data [B,I,N] -> md (flat, GRU input view) and xTpad [B*N, 32] (K-padded)
__global__ void mask_kernel(const float* __restrict__ data,
                            const int* __restrict__ masks,
                            float* __restrict__ md, float* __restrict__ xTpad) {
    int tid = blockIdx.x * blockDim.x + threadIdx.x;
    if (tid >= cB * cI * cN) return;
    int n = tid % cN;
    int rest = tid / cN;
    int i = rest % cI, b = rest / cI;
    float v = masks[tid] ? data[tid] : 0.f;
    md[tid] = v;
    xTpad[((long long)b * cN + n) * 32 + i] = v;
}

__global__ void deg_kernel(const int* __restrict__ ei, float* __restrict__ deg) {
    int t = blockIdx.x * blockDim.x + threadIdx.x;
    if (t < cE)            atomicAdd(&deg[ei[cE + t]], 1.0f);
    else if (t < cE + cN)  atomicAdd(&deg[t - cE], 2.0f);
}

__global__ void dinv_kernel(const float* __restrict__ deg, float* __restrict__ dinv) {
    int t = blockIdx.x * blockDim.x + threadIdx.x;
    if (t < cN) dinv[t] = (deg[t] > 0.f) ? rsqrtf(deg[t]) : 0.f;
}

__global__ void norm_kernel(const int* __restrict__ ei,
                            const float* __restrict__ dinv,
                            float* __restrict__ nrm) {
    int t = blockIdx.x * blockDim.x + threadIdx.x;
    if (t >= cE + cN) return;
    if (t < cE) nrm[t] = dinv[ei[t]] * dinv[ei[cE + t]];
    else { float d = dinv[t - cE]; nrm[t] = 2.f * d * d; }
}

__global__ void gcn_scatter(const float* __restrict__ xw, const int* __restrict__ ei,
                            const float* __restrict__ nrm, float* __restrict__ agg) {
    long long t = (long long)blockIdx.x * blockDim.x + threadIdx.x;
    const long long total = (long long)cB * (cE + cN);
    if (t >= total) return;
    int e = (int)(t % (cE + cN));
    int b = (int)(t / (cE + cN));
    int s, d;
    if (e < cE) { s = ei[e]; d = ei[cE + e]; } else { s = e - cE; d = s; }
    float wgt = nrm[e];
    const float* sp = xw + ((long long)b * cN + s) * cCG;
    float* dp = agg + ((long long)b * cN + d) * cCG;
    #pragma unroll 8
    for (int c = 0; c < cCG; ++c) atomicAdd(&dp[c], wgt * sp[c]);
}

__global__ void bias_relu_kernel(float* __restrict__ h, const float* __restrict__ bias) {
    int tid = blockIdx.x * blockDim.x + threadIdx.x;
    if (tid >= cB * cN * cCG) return;
    h[tid] = fmaxf(h[tid] + bias[tid % cCG], 0.f);
}

// row-wise softmax, 1 wave per row (wave32 shuffle reductions)
__global__ __launch_bounds__(32)
void softmax_rows(float* __restrict__ S, int cols) {
    int lane = threadIdx.x;
    float* p = S + (long long)blockIdx.x * cols;
    float mx = -1e30f;
    for (int c = lane; c < cols; c += 32) mx = fmaxf(mx, p[c]);
    #pragma unroll
    for (int o = 16; o > 0; o >>= 1) mx = fmaxf(mx, __shfl_xor(mx, o, 32));
    float sum = 0.f;
    for (int c = lane; c < cols; c += 32) { float e = __expf(p[c] - mx); p[c] = e; sum += e; }
    #pragma unroll
    for (int o = 16; o > 0; o >>= 1) sum += __shfl_xor(sum, o, 32);
    float inv = 1.f / sum;
    for (int c = lane; c < cols; c += 32) p[c] *= inv;
}

// GRU: single workgroup, hidden state + gate buffers in LDS, 256 sequential steps
__global__ __launch_bounds__(1024)
void gru_kernel(const float* __restrict__ md, const float* __restrict__ wih,
                const float* __restrict__ whh, const float* __restrict__ bih,
                const float* __restrict__ bhh, float* __restrict__ rnn_out) {
    __shared__ float hbuf[cB * cH];       // 2048 f
    __shared__ float gi[cB * 3 * cH];     // 6144 f
    __shared__ float gh[cB * 3 * cH];     // 6144 f
    int tid = threadIdx.x;
    for (int i = tid; i < cB * cH; i += 1024) hbuf[i] = 0.f;
    __syncthreads();
    for (int t = 0; t < cN; ++t) {
        for (int task = tid; task < cB * 192; task += 1024) {
            int b = task / 192, j = task % 192;
            const float* xt = md + (long long)b * (cI * cN) + t * cI;
            float si = bih[j];
            #pragma unroll
            for (int k = 0; k < cI; ++k) si += wih[j * cI + k] * xt[k];
            gi[task] = si;
            float sh = bhh[j];
            const float* hb = hbuf + b * cH;
            #pragma unroll 8
            for (int k = 0; k < cH; ++k) sh += whh[j * cH + k] * hb[k];
            gh[task] = sh;
        }
        __syncthreads();
        for (int task = tid; task < cB * cH; task += 1024) {
            int b = task / cH, j = task % cH;
            float ir = gi[b * 192 + j],        hr = gh[b * 192 + j];
            float iz = gi[b * 192 + 64 + j],   hz = gh[b * 192 + 64 + j];
            float in = gi[b * 192 + 128 + j],  hn = gh[b * 192 + 128 + j];
            float r = 1.f / (1.f + __expf(-(ir + hr)));
            float z = 1.f / (1.f + __expf(-(iz + hz)));
            float nn = tanhf(in + r * hn);
            float hnew = (1.f - z) * nn + z * hbuf[task];
            hbuf[task] = hnew;
            rnn_out[((long long)b * cN + t) * cH + j] = hnew;
        }
        __syncthreads();
    }
}

// embed[b, n*128 + j] = j<64 ? rnn_out[b,n,j] : se[b,n,j-64]
__global__ void embed_kernel(const float* __restrict__ rnn_out,
                             const float* __restrict__ h, float* __restrict__ embed) {
    int tid = blockIdx.x * blockDim.x + threadIdx.x;
    if (tid >= cB * cN * 2 * cH) return;
    int j = tid % 128;
    int rest = tid / 128;
    int n = rest % cN, b = rest / cN;
    float v;
    if (j < cH) {
        v = rnn_out[((long long)b * cN + n) * cH + j];
    } else {
        int f = n * cCG + (j - cH);
        int c2 = f >> 8, n2 = f & 255;
        v = h[((long long)b * cN + n2) * cCG + c2];
    }
    embed[tid] = v;
}

// dst[b][c][n] = src[b][n][c]  (per-batch C x Nn transpose)
__global__ void trans_out_kernel(const float* __restrict__ src, float* __restrict__ dst,
                                 int C, int Nn) {
    long long total = (long long)cB * C * Nn;
    long long tid = (long long)blockIdx.x * blockDim.x + threadIdx.x;
    if (tid >= total) return;
    int n = (int)(tid % Nn);
    long long rest = tid / Nn;
    int c = (int)(rest % C);
    int b = (int)(rest / C);
    dst[tid] = src[((long long)b * Nn + n) * C + c];
}

// ---------------------------------------------------------------------------
// Host side
// ---------------------------------------------------------------------------
static inline void gemm(hipStream_t s, const float* A, const float* Bm, float* C,
                        int M, int N, int K, int batch,
                        long long sA, long long sB, long long sC,
                        int transB, float alpha, const float* bias, int act) {
    dim3 g(N / 64, M / 32, batch);
    if (transB)
        gemm_wmma_kernel<1><<<g, dim3(256), 0, s>>>(A, Bm, C, M, N, K, sA, sB, sC,
                                                    alpha, bias, act);
    else
        gemm_wmma_kernel<0><<<g, dim3(256), 0, s>>>(A, Bm, C, M, N, K, sA, sB, sC,
                                                    alpha, bias, act);
}

extern "C" void kernel_launch(void* const* d_in, const int* in_sizes, int n_in,
                              void* d_out, int out_size, void* d_ws, size_t ws_size,
                              hipStream_t stream) {
    (void)in_sizes; (void)n_in; (void)out_size; (void)ws_size;

    const float* data   = (const float*)d_in[0];
    const int*   masks  = (const int*)d_in[1];
    const int*   ei     = (const int*)d_in[2];
    const float* gcn_w  = (const float*)d_in[3];
    const float* gcn_b  = (const float*)d_in[4];
    const float* sa_qw  = (const float*)d_in[5];
    const float* sa_qb  = (const float*)d_in[6];
    const float* sa_kw  = (const float*)d_in[7];
    const float* sa_kb  = (const float*)d_in[8];
    const float* gwih   = (const float*)d_in[9];
    const float* gwhh   = (const float*)d_in[10];
    const float* gbih   = (const float*)d_in[11];
    const float* gbhh   = (const float*)d_in[12];
    const float* ta_qw  = (const float*)d_in[13];
    const float* ta_qb  = (const float*)d_in[14];
    const float* ta_kw  = (const float*)d_in[15];
    const float* ta_kb  = (const float*)d_in[16];
    const float* fc1_w  = (const float*)d_in[17];
    const float* fc1_b  = (const float*)d_in[18];
    const float* fc2_w  = (const float*)d_in[19];
    const float* fc2_b  = (const float*)d_in[20];

    float* out = (float*)d_out;
    float* out_final = out;                                   // [B,O,N]
    float* out_sa    = out + (long long)cB * cO * cN;         // [B,CG,N]
    float* out_rnna  = out_sa + (long long)cB * cN * cCG;     // [B,H,N]

    // workspace carve-up
    float* w = (float*)d_ws;
    size_t off = 0;
    auto alloc = [&](size_t n) { float* p = w + off; off += n; return p; };
    float* md    = alloc((size_t)cB * cI * cN);
    float* xTp   = alloc((size_t)cB * cN * 32);     // K-padded to 32
    float* gwpad = alloc((size_t)32 * cCG);         // gcn_w padded to K=32
    float* deg   = alloc(cN);
    float* dinv  = alloc(cN);
    float* nrm   = alloc(cE + cN);
    float* xw    = alloc((size_t)cB * cN * cCG);
    float* h     = alloc((size_t)cB * cN * cCG);
    float* q     = alloc((size_t)cB * cN * cCG);
    float* k     = alloc((size_t)cB * cN * cCG);
    float* sc    = alloc((size_t)cB * cN * cN);     // scores (reused temporal)
    float* ha    = alloc((size_t)cB * cN * cCG);
    float* rnn   = alloc((size_t)cB * cN * cH);
    float* qt    = alloc((size_t)cB * cN * cH);
    float* kt    = alloc((size_t)cB * cN * cH);
    float* rnat  = alloc((size_t)cB * cN * cH);
    float* emb   = alloc((size_t)cB * cN * 2 * cH);
    float* inter = alloc((size_t)cB * 4096);
    float* fflat = alloc((size_t)cB * 4096);

    const int TB = 256;
    auto blocks = [](long long n, int tb) { return (unsigned)((n + tb - 1) / tb); };

    // --- mask + padded transpose + padded gcn weights ---
    zero_f32<<<256, 256, 0, stream>>>(xTp, (long long)cB * cN * 32);
    zero_f32<<<8, 256, 0, stream>>>(gwpad, 32 * cCG);
    mask_kernel<<<blocks((long long)cB * cI * cN, TB), TB, 0, stream>>>(data, masks, md, xTp);
    copy_f32<<<blocks(cI * cCG, TB), TB, 0, stream>>>(gcn_w, gwpad, cI * cCG);

    // --- GCN normalization coefficients ---
    zero_f32<<<1, 256, 0, stream>>>(deg, cN);
    deg_kernel<<<blocks(cE + cN, TB), TB, 0, stream>>>(ei, deg);
    dinv_kernel<<<1, 256, 0, stream>>>(deg, dinv);
    norm_kernel<<<blocks(cE + cN, TB), TB, 0, stream>>>(ei, dinv, nrm);

    // --- xw = xTpad @ gwpad  [8192,32]x[32,64] ---
    gemm(stream, xTp, gwpad, xw, cB * cN, cCG, 32, 1, 0, 0, 0, 0, 1.f, nullptr, 0);

    // --- aggregate + bias + relu ---
    zero_f32<<<1024, 256, 0, stream>>>(h, (long long)cB * cN * cCG);
    gcn_scatter<<<blocks((long long)cB * (cE + cN), TB), TB, 0, stream>>>(xw, ei, nrm, h);
    bias_relu_kernel<<<blocks((long long)cB * cN * cCG, TB), TB, 0, stream>>>(h, gcn_b);

    // --- spatial attention ---
    gemm(stream, h, sa_qw, q, cB * cN, cCG, cCG, 1, 0, 0, 0, 0, 1.f, sa_qb, 0);
    gemm(stream, h, sa_kw, k, cB * cN, cCG, cCG, 1, 0, 0, 0, 0, 1.f, sa_kb, 0);
    gemm(stream, q, k, sc, cN, cN, cCG, cB,
         (long long)cN * cCG, (long long)cN * cCG, (long long)cN * cN,
         1, 0.0625f, nullptr, 0);                         // 1/sqrt(256)
    softmax_rows<<<cB * cN, 32, 0, stream>>>(sc, cN);
    gemm(stream, sc, h, ha, cN, cCG, cN, cB,
         (long long)cN * cN, (long long)cN * cCG, (long long)cN * cCG,
         0, 1.f, nullptr, 0);
    trans_out_kernel<<<blocks((long long)cB * cCG * cN, TB), TB, 0, stream>>>(ha, out_sa, cCG, cN);

    // --- GRU ---
    gru_kernel<<<1, 1024, 0, stream>>>(md, gwih, gwhh, gbih, gbhh, rnn);

    // --- temporal attention ---
    gemm(stream, rnn, ta_qw, qt, cB * cN, cH, cH, 1, 0, 0, 0, 0, 1.f, ta_qb, 0);
    gemm(stream, rnn, ta_kw, kt, cB * cN, cH, cH, 1, 0, 0, 0, 0, 1.f, ta_kb, 0);
    gemm(stream, qt, kt, sc, cN, cN, cH, cB,
         (long long)cN * cH, (long long)cN * cH, (long long)cN * cN,
         1, 0.0625f, nullptr, 0);                         // 1/sqrt(SEQ_LEN=256)
    softmax_rows<<<cB * cN, 32, 0, stream>>>(sc, cN);
    gemm(stream, sc, rnn, rnat, cN, cH, cN, cB,
         (long long)cN * cN, (long long)cN * cH, (long long)cN * cH,
         0, 1.f, nullptr, 0);
    trans_out_kernel<<<blocks((long long)cB * cH * cN, TB), TB, 0, stream>>>(rnat, out_rnna, cH, cN);

    // --- fused MLP ---
    embed_kernel<<<blocks((long long)cB * cN * 2 * cH, TB), TB, 0, stream>>>(rnn, h, emb);
    gemm(stream, emb, fc1_w, inter, cB, 4096, 2 * cH * cN, 1, 0, 0, 0, 0, 1.f, fc1_b, 0);
    gemm(stream, inter, fc2_w, fflat, cB, 4096, 4096, 1, 0, 0, 0, 0, 1.f, fc2_b, 0);
    trans_out_kernel<<<blocks((long long)cB * cO * cN, TB), TB, 0, stream>>>(fflat, out_final, cO, cN);
}
